// Sae_85985245266007
// MI455X (gfx1250) — matrix operand order repaired and checked by reading, not compiled, not run on hardware
//
#include <hip/hip_runtime.h>

#define N_TOK 4096
#define D_INN 2048
#define D_SAE 32768
#define TOPK  32
#define MROWS 32            // token rows per block (2 WMMA row-panels share every B fragment)

typedef __attribute__((ext_vector_type(16))) __bf16 v16bf;
typedef __attribute__((ext_vector_type(8)))  __bf16 v8bf;
typedef __attribute__((ext_vector_type(8)))  float  v8f;

__device__ __forceinline__ unsigned short f32_bf16(float f) {
  unsigned u = __float_as_uint(f);
  u += 0x7FFFu + ((u >> 16) & 1u);   // round-to-nearest-even
  return (unsigned short)(u >> 16);
}
__device__ __forceinline__ unsigned pack2(float lo, float hi) {
  return (unsigned)f32_bf16(lo) | ((unsigned)f32_bf16(hi) << 16);
}

// ---------------- conversion kernels ----------------
__global__ void k_conv_x(const float* __restrict__ x, const float* __restrict__ b_dec,
                         unsigned short* __restrict__ xb) {
  long long t = (long long)blockIdx.x * blockDim.x + threadIdx.x;
  long long base = t * 8;
  int col = (int)(base & (D_INN - 1));
  float4 a0 = *(const float4*)(x + base);
  float4 a1 = *(const float4*)(x + base + 4);
  float4 b0 = *(const float4*)(b_dec + col);
  float4 b1 = *(const float4*)(b_dec + col + 4);
  uint4 o;
  o.x = pack2(a0.x - b0.x, a0.y - b0.y);
  o.y = pack2(a0.z - b0.z, a0.w - b0.w);
  o.z = pack2(a1.x - b1.x, a1.y - b1.y);
  o.w = pack2(a1.z - b1.z, a1.w - b1.w);
  *(uint4*)(xb + base) = o;
}

__global__ void k_conv_w(const float* __restrict__ w, unsigned short* __restrict__ wb) {
  long long t = (long long)blockIdx.x * blockDim.x + threadIdx.x;
  long long base = t * 8;
  float4 a0 = *(const float4*)(w + base);
  float4 a1 = *(const float4*)(w + base + 4);
  uint4 o;
  o.x = pack2(a0.x, a0.y);
  o.y = pack2(a0.z, a0.w);
  o.z = pack2(a1.x, a1.y);
  o.w = pack2(a1.z, a1.w);
  *(uint4*)(wb + base) = o;
}

// ---------------- column variance ----------------
__global__ void k_stats(const float* __restrict__ x, float* __restrict__ tv) {
  int c = blockIdx.x * 256 + threadIdx.x;
  float s = 0.f, s2 = 0.f;
  for (int n = 0; n < N_TOK; ++n) {
    float v = x[(long long)n * D_INN + c];
    s += v; s2 += v * v;
  }
  tv[c] = s2 - s * s * (1.0f / (float)N_TOK);
}

// ---------------- main: encoder WMMA GEMM + streaming top-k + decode ----------------
__launch_bounds__(256)
__global__ void k_main(const unsigned short* __restrict__ xb,
                       const unsigned short* __restrict__ wb,
                       const float* __restrict__ enc_b,
                       const float* __restrict__ W_dec,
                       const float* __restrict__ b_dec,
                       const float* __restrict__ x,
                       float* __restrict__ sae_out,
                       float* __restrict__ top_acts,
                       float* __restrict__ top_idx,
                       float* __restrict__ l2p) {
  extern __shared__ char smem[];
  unsigned short* sA = (unsigned short*)smem;                 // 32 x 2048 bf16 = 128 KB
  float* lvals  = (float*)(smem + 131072);                    // 8 waves x 32 rows x 32 = 32 KB
  int*   lidx   = (int*)  (smem + 131072 + 32768);            // 32 KB
  float* rowmin = (float*)(smem + 131072 + 65536);            // 8 x 32 = 1 KB
  float* fActs  = (float*)(smem + 131072 + 65536 + 1024);     // 32 x 32 = 4 KB
  int*   fIdx   = (int*)  (smem + 131072 + 65536 + 1024 + 4096);

  const int tid  = threadIdx.x;
  const int wave = tid >> 5;
  const int lane = tid & 31;
  const int hi   = lane >> 4;
  const int l16  = lane & 15;
  const int row0 = blockIdx.x * MROWS;

  // Stage 32-row A-panel into LDS once; reused for all 256 column tiles.
  {
    const uint4* src = (const uint4*)(xb + (long long)row0 * D_INN);
    uint4* dst = (uint4*)sA;
    for (int i = tid; i < (MROWS * D_INN * 2) / 16; i += 256) dst[i] = src[i];
  }
  for (int i = tid; i < 8 * MROWS * 32; i += 256) { lvals[i] = -__builtin_inff(); lidx[i] = 0; }
  rowmin[tid] = -__builtin_inff();   // exactly 8*32 = 256 entries
  __syncthreads();

  const unsigned short* arow0 = sA + l16 * D_INN + hi * 8;          // panel 0: rows 0..15
  const unsigned short* arow1 = sA + (16 + l16) * D_INN + hi * 8;   // panel 1: rows 16..31

  for (int tile = 0; tile < D_SAE / 128; ++tile) {
    const int colg = tile * 128 + wave * 16 + l16;
    const unsigned short* bcol = wb + (long long)colg * D_INN + hi * 8;
    v8f c0 = {}, c1 = {};

    // Software-pipelined K loop: fragments for step kk+32 are in flight while
    // step kk's WMMA pair executes (1 block/WGP -> rely on ILP, not TLP).
    v16bf bv = __builtin_shufflevector(*(const v8bf*)(bcol), *(const v8bf*)(bcol + 16), 0,1,2,3,4,5,6,7,8,9,10,11,12,13,14,15);
    v8bf t0a = *(const v8bf*)(arow0);
    v8bf t0b = *(const v8bf*)(arow0 + 16);
    v8bf t1a = *(const v8bf*)(arow1);
    v8bf t1b = *(const v8bf*)(arow1 + 16);
    v16bf av0 = __builtin_shufflevector(t0a, t0b, 0,1,2,3,4,5,6,7,8,9,10,11,12,13,14,15);
    v16bf av1 = __builtin_shufflevector(t1a, t1b, 0,1,2,3,4,5,6,7,8,9,10,11,12,13,14,15);
#pragma unroll 4
    for (int kk = 32; kk < D_INN; kk += 32) {
      v16bf bn  = __builtin_shufflevector(*(const v8bf*)(bcol + kk), *(const v8bf*)(bcol + kk + 16), 0,1,2,3,4,5,6,7,8,9,10,11,12,13,14,15);
      v8bf  n0a = *(const v8bf*)(arow0 + kk);
      v8bf  n0b = *(const v8bf*)(arow0 + kk + 16);
      v8bf  n1a = *(const v8bf*)(arow1 + kk);
      v8bf  n1b = *(const v8bf*)(arow1 + kk + 16);
      if ((kk & 127) == 32)   // once per 128 K: pre-warm next tile's B stream from L2
        __builtin_prefetch(bcol + (long long)128 * D_INN + kk, 0, 1);
      c0 = __builtin_amdgcn_wmma_f32_16x16x32_bf16(false, av0, false, bv, (short)0, c0,
                                                   false, false);
      c1 = __builtin_amdgcn_wmma_f32_16x16x32_bf16(false, av1, false, bv, (short)0, c1,
                                                   false, false);
      bv  = bn;
      av0 = __builtin_shufflevector(n0a, n0b, 0,1,2,3,4,5,6,7,8,9,10,11,12,13,14,15);
      av1 = __builtin_shufflevector(n1a, n1b, 0,1,2,3,4,5,6,7,8,9,10,11,12,13,14,15);
    }
    c0 = __builtin_amdgcn_wmma_f32_16x16x32_bf16(false, av0, false, bv, (short)0, c0,
                                                 false, false);
    c1 = __builtin_amdgcn_wmma_f32_16x16x32_bf16(false, av1, false, bv, (short)0, c1,
                                                 false, false);

    const float bias = enc_b[colg];
#pragma unroll
    for (int p = 0; p < 2; ++p) {
#pragma unroll
      for (int v = 0; v < 8; ++v) {
        int row = p * 16 + (hi ? (v + 8) : v);
        float val = (p ? c1[v] : c0[v]) + bias;
        val = val > 0.f ? val : 0.f;                  // relu
        float rmin = rowmin[wave * MROWS + row];
        unsigned long long m = __ballot(val > rmin);  // only candidates beating running min
        while (m) {
          int src = __ffsll(m) - 1;
          m &= (m - 1);
          float iv  = __shfl(val, src);
          int  irow = p * 16 + ((src >> 4) ? (v + 8) : v);
          int  icol = tile * 128 + wave * 16 + (src & 15);
          int  lr   = wave * MROWS + irow;
          float* lv = lvals + lr * 32;
          int*   li = lidx  + lr * 32;
          // wave-cooperative argmin-replace
          float cur = lv[lane];
          float mv = cur; int ml = lane;
#pragma unroll
          for (int off = 16; off; off >>= 1) {
            float ov = __shfl_xor(mv, off);
            int   ol = __shfl_xor(ml, off);
            if (ov < mv || (ov == mv && ol < ml)) { mv = ov; ml = ol; }
          }
          if (iv > mv) {
            if (lane == ml) { lv[lane] = iv; li[lane] = icol; cur = iv; }
            float nv = cur;
#pragma unroll
            for (int off = 16; off; off >>= 1) nv = fminf(nv, __shfl_xor(nv, off));
            if (lane == 0) rowmin[lr] = nv;
          }
        }
      }
    }
  }
  __syncthreads();

  // Merge 8 per-wave lists (256 candidates) -> exact top-32 per row, descending.
  for (int r = wave * 4; r < wave * 4 + 4; ++r) {
    float cv[8]; int ci[8];
#pragma unroll
    for (int j = 0; j < 8; ++j) {
      int a = (j * MROWS + r) * 32 + lane;
      cv[j] = lvals[a]; ci[j] = lidx[a];
    }
    const long long rowg = row0 + r;
    for (int sel = 0; sel < TOPK; ++sel) {
      float bv = cv[0]; int bj = 0;
#pragma unroll
      for (int j = 1; j < 8; ++j) if (cv[j] > bv) { bv = cv[j]; bj = j; }
      float mv = bv; int ml = lane;
#pragma unroll
      for (int off = 16; off; off >>= 1) {
        float ov = __shfl_xor(mv, off);
        int   ol = __shfl_xor(ml, off);
        if (ov > mv || (ov == mv && ol < ml)) { mv = ov; ml = ol; }
      }
      int wcol = __shfl(ci[bj], ml);
      if (lane == ml) cv[bj] = -__builtin_inff();
      float outv = mv > 0.f ? mv : 0.f;
      if (lane == 0) {
        fActs[r * 32 + sel] = outv;
        fIdx [r * 32 + sel] = wcol;
        top_acts[rowg * TOPK + sel] = outv;
        top_idx [rowg * TOPK + sel] = (float)wcol;
      }
    }
  }
  __syncthreads();

  // Sparse decode + per-block l2 partials (deterministic, no atomics).
  const int c0i = tid * 8;
  float l2a[8];
#pragma unroll
  for (int j = 0; j < 8; ++j) l2a[j] = 0.f;
  for (int r = 0; r < MROWS; ++r) {
    float acc[8];
#pragma unroll
    for (int j = 0; j < 8; ++j) acc[j] = 0.f;
    for (int kk = 0; kk < TOPK; ++kk) {
      float a  = fActs[r * 32 + kk];
      int   id = fIdx [r * 32 + kk];
      const float4* wr = (const float4*)(W_dec + (long long)id * D_INN + c0i);
      float4 w0 = wr[0], w1 = wr[1];
      acc[0] += a * w0.x; acc[1] += a * w0.y; acc[2] += a * w0.z; acc[3] += a * w0.w;
      acc[4] += a * w1.x; acc[5] += a * w1.y; acc[6] += a * w1.z; acc[7] += a * w1.w;
    }
    const long long rowg = row0 + r;
    const float4* bd = (const float4*)(b_dec + c0i);
    const float4* xr = (const float4*)(x + rowg * D_INN + c0i);
    float4 b0 = bd[0], b1 = bd[1], x0 = xr[0], x1 = xr[1];
    float o[8];
    o[0] = acc[0] + b0.x; o[1] = acc[1] + b0.y; o[2] = acc[2] + b0.z; o[3] = acc[3] + b0.w;
    o[4] = acc[4] + b1.x; o[5] = acc[5] + b1.y; o[6] = acc[6] + b1.z; o[7] = acc[7] + b1.w;
    float4 s0 = { o[0], o[1], o[2], o[3] };
    float4 s1 = { o[4], o[5], o[6], o[7] };
    *(float4*)(sae_out + rowg * D_INN + c0i)     = s0;
    *(float4*)(sae_out + rowg * D_INN + c0i + 4) = s1;
    float e0 = o[0]-x0.x, e1 = o[1]-x0.y, e2 = o[2]-x0.z, e3 = o[3]-x0.w;
    float e4 = o[4]-x1.x, e5 = o[5]-x1.y, e6 = o[6]-x1.z, e7 = o[7]-x1.w;
    l2a[0]+=e0*e0; l2a[1]+=e1*e1; l2a[2]+=e2*e2; l2a[3]+=e3*e3;
    l2a[4]+=e4*e4; l2a[5]+=e5*e5; l2a[6]+=e6*e6; l2a[7]+=e7*e7;
  }
  float* dst = l2p + (long long)blockIdx.x * D_INN + c0i;
#pragma unroll
  for (int j = 0; j < 8; ++j) dst[j] = l2a[j];
}

// ---------------- fvu + auxk ----------------
__global__ void k_fvu(const float* __restrict__ l2p, const float* __restrict__ tv,
                      float* __restrict__ out) {
  __shared__ float red[256];
  float s = 0.f;
  for (int c = threadIdx.x; c < D_INN; c += 256) {
    float l2c = 0.f;
    for (int b = 0; b < N_TOK / MROWS; ++b) l2c += l2p[(long long)b * D_INN + c];
    s += l2c / tv[c];
  }
  red[threadIdx.x] = s;
  __syncthreads();
  for (int o = 128; o; o >>= 1) {
    if (threadIdx.x < o) red[threadIdx.x] += red[threadIdx.x + o];
    __syncthreads();
  }
  if (threadIdx.x == 0) {
    out[0] = red[0] * (1.0f / (float)D_INN);  // fvu
    out[1] = 0.f;                             // auxk_loss
  }
}

extern "C" void kernel_launch(void* const* d_in, const int* in_sizes, int n_in,
                              void* d_out, int out_size, void* d_ws, size_t ws_size,
                              hipStream_t stream) {
  (void)in_sizes; (void)n_in; (void)out_size; (void)ws_size;
  const float* x     = (const float*)d_in[0];
  const float* enc_W = (const float*)d_in[1];
  const float* enc_b = (const float*)d_in[2];
  const float* W_dec = (const float*)d_in[3];
  const float* b_dec = (const float*)d_in[4];

  float* out      = (float*)d_out;
  float* sae_out  = out;                                    // 4096*2048
  float* top_acts = out + (long long)N_TOK * D_INN;         // 4096*32
  float* top_idx  = top_acts + (long long)N_TOK * TOPK;     // 4096*32 (as float)
  float* fvu_out  = top_idx + (long long)N_TOK * TOPK;      // [fvu, auxk]

  char* ws = (char*)d_ws;
  unsigned short* xb = (unsigned short*)ws;                                  // 16 MB
  unsigned short* wb = (unsigned short*)(ws + (size_t)N_TOK * D_INN * 2);    // 128 MB (fits L2)
  float* tv  = (float*)(ws + (size_t)N_TOK * D_INN * 2 + (size_t)D_SAE * D_INN * 2);
  float* l2p = tv + D_INN;                                                   // 128*2048

  k_conv_x<<<(N_TOK * D_INN) / (256 * 8), 256, 0, stream>>>(x, b_dec, xb);
  k_conv_w<<<(int)(((long long)D_SAE * D_INN) / (256 * 8)), 256, 0, stream>>>(enc_W, wb);
  k_stats<<<D_INN / 256, 256, 0, stream>>>(x, tv);

  size_t smem = 131072 + 32768 + 32768 + 1024 + 4096 + 4096;  // ~200.7 KB of 320 KB LDS
  (void)hipFuncSetAttribute((const void*)k_main,
                            hipFuncAttributeMaxDynamicSharedMemorySize, (int)smem);
  k_main<<<N_TOK / MROWS, 256, smem, stream>>>(xb, wb, enc_b, W_dec, b_dec, x,
                                               sae_out, top_acts, top_idx, l2p);
  k_fvu<<<1, 256, 0, stream>>>(l2p, tv, fvu_out);
}
